// Router_10900626997977
// MI455X (gfx1250) — compile-verified, matrix operand order checked by reading, not run on hardware
//
#include <hip/hip_runtime.h>
#include <hip/hip_bf16.h>

// ---------------------------------------------------------------------------
// MoE router for MI455X (gfx1250, wave32).
//   logits GEMM  : v_wmma_f32_16x16x32_f16 (f16 mul, f32 acc) - memory bound,
//                  512 MiB of x @ 23.3 TB/s ~ 22 us lower bound. K-loop is
//                  unrolled x4 so each wave keeps up to 32 b128 loads in
//                  flight and issues 4 WMMAs back-to-back per trip.
//   softmax/argmax/l_aux stats fused into the GEMM epilogue.
//   per-expert top-2048: histogram select + in-LDS bitonic sort.
//   new_mask1: exact stable per-expert rank (chunk count -> scan -> emit).
// ---------------------------------------------------------------------------

#define S_TOK   131072
#define M_DIM   1024
#define E_EXP   64
#define CAP     2048
#define NBKT    1024          // bucket = float_bits >> 20 (exp + 3 mantissa bits)
#define NCHUNK  (S_TOK / CAP) // 64 chunks of 2048 tokens

typedef __attribute__((ext_vector_type(16))) _Float16 v16h;
typedef __attribute__((ext_vector_type(8)))  float    v8f;
typedef __attribute__((ext_vector_type(4)))  float    v4f;

// ---------------------------------------------------------------------------
// K1: 16 tokens x 64 experts per block (128 thr = 4 waves, one 16x16 C tile
// per wave). K step of 32 per WMMA, unrolled x4. Epilogue: softmax, argmax,
// me/ce stats, gate histogram for the capacity threshold.
// ---------------------------------------------------------------------------
__global__ __launch_bounds__(128)
void k_gemm_softmax(const float* __restrict__ x, const float* __restrict__ wg,
                    float* __restrict__ gates, unsigned* __restrict__ eid,
                    float* __restrict__ me, unsigned* __restrict__ ce,
                    unsigned* __restrict__ hist) {
  __shared__ float lds[16 * 68];  // 16 tokens x 64 experts (+4 pad)
  __shared__ float lme[E_EXP];

  const int tid  = threadIdx.x;
  const int lane = tid & 31;
  const int wave = tid >> 5;
  const int t0   = blockIdx.x * 16;
  const int row  = lane & 15;
  const int hi   = lane >> 4;       // 0: lanes 0-15, 1: lanes 16-31
  const int e0   = wave * 16;

  // 16-bit A/B fragment layout (ISA 7.12.2): lanes 0-15 hold K = {o0..o0+7,
  // o1..o1+7} with o0=0,o1=16; lanes 16-31 hold o0=8,o1=24.
  const int o0 = hi ? 8 : 0;
  const int o1 = hi ? 24 : 16;
  const float* xr = x  + (size_t)(t0 + row) * M_DIM;
  const float* wr = wg + (size_t)(e0 + row) * M_DIM;

  v8f acc = {};
#pragma unroll 4
  for (int k0 = 0; k0 < M_DIM; k0 += 32) {
    __builtin_prefetch(xr + k0 + 256, 0, 1);  // global_prefetch_b8, next slices
    v4f a0 = *(const v4f*)(xr + k0 + o0);
    v4f a1 = *(const v4f*)(xr + k0 + o0 + 4);
    v4f a2 = *(const v4f*)(xr + k0 + o1);
    v4f a3 = *(const v4f*)(xr + k0 + o1 + 4);
    v4f b0 = *(const v4f*)(wr + k0 + o0);
    v4f b1 = *(const v4f*)(wr + k0 + o0 + 4);
    v4f b2 = *(const v4f*)(wr + k0 + o1);
    v4f b3 = *(const v4f*)(wr + k0 + o1 + 4);
    v16h A = {}, B = {};
#pragma unroll
    for (int i = 0; i < 4; ++i) {
      A[i]      = (_Float16)a0[i];
      A[i + 4]  = (_Float16)a1[i];
      A[i + 8]  = (_Float16)a2[i];
      A[i + 12] = (_Float16)a3[i];
      B[i]      = (_Float16)b0[i];
      B[i + 4]  = (_Float16)b1[i];
      B[i + 8]  = (_Float16)b2[i];
      B[i + 12] = (_Float16)b3[i];
    }
    acc = __builtin_amdgcn_wmma_f32_16x16x32_f16(
        false, A, false, B, (short)0, acc, false, false);
  }

  // C layout: VGPR i -> M = i (lanes 0-15) / i+8 (lanes 16-31), N = lane&15.
#pragma unroll
  for (int i = 0; i < 8; ++i) {
    int r = hi ? (i + 8) : i;
    lds[r * 68 + e0 + row] = acc[i];
  }
  if (tid < E_EXP) lme[tid] = 0.f;
  __syncthreads();

  // Softmax + argmax, one thread per token (64 experts serial: trivial cost).
  if (tid < 16) {
    float* L = &lds[tid * 68];
    float mx = L[0]; int am = 0;
    for (int j = 1; j < E_EXP; ++j) { float v = L[j]; if (v > mx) { mx = v; am = j; } }
    float sum = 0.f;
    for (int j = 0; j < E_EXP; ++j) { float v = __expf(L[j] - mx); L[j] = v; sum += v; }
    float inv = 1.f / sum;
    for (int j = 0; j < E_EXP; ++j) L[j] *= inv;
    eid[t0 + tid] = (unsigned)am;
    atomicAdd(&ce[am], 1u);
  }
  __syncthreads();

  // Emit gates, histogram buckets, per-block me partials (LDS ds_add_f32).
  for (int i = tid; i < 16 * E_EXP; i += 128) {
    int r = i >> 6, c = i & 63;
    float g = lds[r * 68 + c];
    gates[(size_t)(t0 + r) * E_EXP + c] = g;
    unsigned b = __float_as_uint(g) >> 20;   // gates in [0,1] -> b <= 1016
    if (b > NBKT - 1) b = NBKT - 1;
    atomicAdd(&hist[c * NBKT + b], 1u);
    atomicAdd(&lme[c], g);
  }
  __syncthreads();
  if (tid < E_EXP) atomicAdd(&me[tid], lme[tid]);
}

// ---------------------------------------------------------------------------
// K2: per-expert suffix scan of the histogram -> threshold bucket holding the
// capacity-th largest gate. Block 0 also folds me/ce into l_aux.
// ---------------------------------------------------------------------------
__global__ __launch_bounds__(256)
void k_thresh(const unsigned* __restrict__ hist, unsigned* __restrict__ thresh,
              const float* __restrict__ me, const unsigned* __restrict__ ce,
              float* __restrict__ out) {
  __shared__ unsigned h[NBKT];
  int e = blockIdx.x;
  for (int i = threadIdx.x; i < NBKT; i += 256) h[i] = hist[e * NBKT + i];
  __syncthreads();
  if (threadIdx.x == 0) {
    unsigned acc = 0; int b = NBKT - 1;
    for (; b > 0; --b) { acc += h[b]; if (acc >= CAP) break; }
    thresh[e] = (unsigned)b;
  }
  if (blockIdx.x == 0 && threadIdx.x == 1) {
    float l = 0.f;
    for (int j = 0; j < E_EXP; ++j)
      l += (me[j] / (float)S_TOK) * ((float)ce[j] / (float)S_TOK);
    out[0] = l * (float)E_EXP;
  }
}

// ---------------------------------------------------------------------------
// K3: threshold compaction of (gate, token) candidates per expert.
// Every expert has >= CAP entries at >= threshold bucket, so all CAP slots
// fill (ties inside the threshold bucket resolved by atomic arrival order).
// ---------------------------------------------------------------------------
__global__ __launch_bounds__(256)
void k_compact(const float* __restrict__ gates, const unsigned* __restrict__ thresh,
               unsigned* __restrict__ cnt, float* __restrict__ cval,
               int* __restrict__ cidx) {
  size_t f = (size_t)blockIdx.x * 256 + threadIdx.x;   // token*64 + expert
  int t = (int)(f >> 6), e = (int)(f & 63);
  float g = gates[f];
  unsigned b = __float_as_uint(g) >> 20;
  if (b > NBKT - 1) b = NBKT - 1;
  if (b >= thresh[e]) {
    unsigned pos = atomicAdd(&cnt[e], 1u);
    if (pos < CAP) { cval[e * CAP + pos] = g; cidx[e * CAP + pos] = t; }
  }
}

// ---------------------------------------------------------------------------
// K4a/b/c: exact stable argmax-rank for new_mask1.
// ---------------------------------------------------------------------------
__global__ __launch_bounds__(64)
void k_chunk_count(const unsigned* __restrict__ eid, unsigned* __restrict__ chunkCnt) {
  int chunk = blockIdx.x, e = threadIdx.x, base = chunk * CAP;
  unsigned c = 0;
  for (int i = 0; i < CAP; ++i) c += (eid[base + i] == (unsigned)e) ? 1u : 0u;
  chunkCnt[chunk * E_EXP + e] = c;
}

__global__ __launch_bounds__(64)
void k_chunk_scan(const unsigned* __restrict__ chunkCnt, unsigned* __restrict__ chunkBase) {
  int e = threadIdx.x; unsigned base = 0;
  for (int c = 0; c < NCHUNK; ++c) {
    chunkBase[c * E_EXP + e] = base;
    base += chunkCnt[c * E_EXP + e];
  }
}

__global__ __launch_bounds__(64)
void k_mask(const unsigned* __restrict__ eid, const unsigned* __restrict__ chunkBase,
            float* __restrict__ mask_out) {
  int chunk = blockIdx.x, e = threadIdx.x, base = chunk * CAP;
  unsigned r = chunkBase[chunk * E_EXP + e];
  for (int i = 0; i < CAP; ++i) {
    if (eid[base + i] == (unsigned)e) {
      if (r < CAP) mask_out[(size_t)(base + i) * E_EXP + e] = 1.0f;
      ++r;
    }
  }
}

// ---------------------------------------------------------------------------
// K5: per-expert in-LDS bitonic sort of the 2048 candidates, descending by
// gate with ascending-token-index tie break; emits gatings + indices.
// ---------------------------------------------------------------------------
__global__ __launch_bounds__(1024)
void k_sort(const float* __restrict__ cval, const int* __restrict__ cidx,
            float* __restrict__ gat_out, int* __restrict__ idx_out) {
  __shared__ float sv[CAP];
  __shared__ int   si[CAP];
  int e = blockIdx.x;
  for (int i = threadIdx.x; i < CAP; i += 1024) {
    sv[i] = cval[e * CAP + i];
    si[i] = cidx[e * CAP + i];
  }
  __syncthreads();
  for (int k = 2; k <= CAP; k <<= 1) {
    for (int j = k >> 1; j > 0; j >>= 1) {
      for (int i = threadIdx.x; i < CAP; i += 1024) {
        int l = i ^ j;
        if (l > i) {
          float av = sv[i], bv = sv[l];
          int   ai = si[i], bi = si[l];
          bool pre  = (av > bv) || (av == bv && ai < bi);  // i should precede l
          bool desc = ((i & k) == 0);
          if (desc ? !pre : pre) { sv[i] = bv; si[i] = bi; sv[l] = av; si[l] = ai; }
        }
      }
      __syncthreads();
    }
  }
  for (int i = threadIdx.x; i < CAP; i += 1024) {
    gat_out[e * CAP + i] = sv[i];
    idx_out[e * CAP + i] = si[i];
  }
}

// ---------------------------------------------------------------------------
extern "C" void kernel_launch(void* const* d_in, const int* in_sizes, int n_in,
                              void* d_out, int out_size, void* d_ws, size_t ws_size,
                              hipStream_t stream) {
  (void)in_sizes; (void)n_in; (void)ws_size;
  const float* x  = (const float*)d_in[0];   // [131072, 1024] f32
  const float* wg = (const float*)d_in[1];   // [64, 1024]     f32

  // Output layout (flat, reference return order):
  //   [0]                      l_aux
  //   [1 .. 1+64*2048)         gatings  [e, cap] f32
  //   next 64*2048             indices  [e, cap] int32 (bit-stored)
  //   next 131072*64           new_mask1 [s, e] f32
  float* out      = (float*)d_out;
  float* out_gat  = out + 1;
  int*   out_idx  = (int*)(out + 1 + E_EXP * CAP);
  float* out_mask = out + 1 + 2 * E_EXP * CAP;

  // Workspace carve-up (256B aligned slices).
  char* w = (char*)d_ws;
  auto alloc = [&](size_t bytes) {
    char* p = w; w += (bytes + 255) & ~(size_t)255; return p;
  };
  float*    gates     = (float*)   alloc((size_t)S_TOK * E_EXP * 4);  // 33.5 MB
  unsigned* eid       = (unsigned*)alloc((size_t)S_TOK * 4);
  float*    me        = (float*)   alloc(E_EXP * 4);
  unsigned* ce        = (unsigned*)alloc(E_EXP * 4);
  unsigned* hist      = (unsigned*)alloc((size_t)E_EXP * NBKT * 4);
  unsigned* thresh    = (unsigned*)alloc(E_EXP * 4);
  unsigned* cnt       = (unsigned*)alloc(E_EXP * 4);
  unsigned* chunkCnt  = (unsigned*)alloc((size_t)NCHUNK * E_EXP * 4);
  unsigned* chunkBase = (unsigned*)alloc((size_t)NCHUNK * E_EXP * 4);
  float*    cval      = (float*)   alloc((size_t)E_EXP * CAP * 4);
  int*      cidx      = (int*)     alloc((size_t)E_EXP * CAP * 4);

  // Deterministic per-call init (graph-capture safe).
  hipMemsetAsync(d_out, 0, (size_t)out_size * sizeof(float), stream);
  hipMemsetAsync(me,   0, E_EXP * 4, stream);
  hipMemsetAsync(ce,   0, E_EXP * 4, stream);
  hipMemsetAsync(cnt,  0, E_EXP * 4, stream);
  hipMemsetAsync(hist, 0, (size_t)E_EXP * NBKT * 4, stream);

  k_gemm_softmax<<<S_TOK / 16, 128, 0, stream>>>(x, wg, gates, eid, me, ce, hist);
  k_thresh     <<<E_EXP, 256, 0, stream>>>(hist, thresh, me, ce, out);
  k_compact    <<<(S_TOK * E_EXP) / 256, 256, 0, stream>>>(gates, thresh, cnt, cval, cidx);
  k_chunk_count<<<NCHUNK, E_EXP, 0, stream>>>(eid, chunkCnt);
  k_chunk_scan <<<1, E_EXP, 0, stream>>>(chunkCnt, chunkBase);
  k_mask       <<<NCHUNK, E_EXP, 0, stream>>>(eid, chunkBase, out_mask);
  k_sort       <<<E_EXP, 1024, 0, stream>>>(cval, cidx, out_gat, out_idx);
}
